// ImageAttn_6725918786182
// MI455X (gfx1250) — matrix operand order repaired and checked by reading, not compiled, not run on hardware
//
#include <hip/hip_runtime.h>

typedef __attribute__((ext_vector_type(16))) _Float16 v16h;
typedef __attribute__((ext_vector_type(8)))  float    v8f;

#define DEV static __device__ __forceinline__

// Problem constants (fixed by the reference)
constexpr int BB   = 2;
constexpr int CC   = 512;     // IN_DIM
constexpr int SS   = 4096;    // 64*64
constexpr int MM   = BB * SS; // 8192 rows
constexpr int KDIM = 256;     // q/k width
constexpr int VDIM = 512;     // v width
constexpr int NH   = 8;
constexpr int DK   = 32;      // per-head q/k
constexpr int DV   = 64;      // per-head v

union Frag16 {            // one WMMA f16 operand: 8 VGPRs / 16 halfs per lane
    v16h  h;
    uint4 q[2];
    unsigned u[8];
};

DEV uint4 ldg4(const _Float16* p) { return *reinterpret_cast<const uint4*>(p); }

// ---- swap data between the two 16-lane halves (lane i <-> lane i^16) ------
#if __has_builtin(__builtin_amdgcn_permlanex16)
DEV float swap16(float x) {
    int r = __builtin_amdgcn_permlanex16(__float_as_int(x), __float_as_int(x),
                                         0x76543210u, 0xFEDCBA98u, false,
                                         false);
    return __int_as_float(r);
}
#else
DEV float swap16(float x) { return __shfl_xor(x, 16, 32); }
#endif

// ---- pack two f32 into one dword of f16 -----------------------------------
#if __has_builtin(__builtin_amdgcn_cvt_pkrtz)
DEV unsigned pk2(float a, float b) {
    return __builtin_bit_cast(unsigned, __builtin_amdgcn_cvt_pkrtz(a, b));
}
#else
DEV unsigned pk2(float a, float b) {
    union { _Float16 h[2]; unsigned u; } c;
    c.h[0] = (_Float16)a; c.h[1] = (_Float16)b;
    return c.u;
}
#endif

// ---- balanced (latency-friendly) register reductions ----------------------
DEV float max8(const v8f& c) {
    float a0 = fmaxf(c[0], c[1]), a1 = fmaxf(c[2], c[3]);
    float a2 = fmaxf(c[4], c[5]), a3 = fmaxf(c[6], c[7]);
    return fmaxf(fmaxf(a0, a1), fmaxf(a2, a3));
}
DEV float sum8(const float (&a)[8]) {
    return ((a[0] + a[1]) + (a[2] + a[3])) + ((a[4] + a[5]) + (a[6] + a[7]));
}

// ---------------------------------------------------------------------------
// Tiled transpose + f32->f16 convert:  in [K][N] f32 -> out [N][K] f16.
// grid = (N/32, K/32, batches), block = (32, 8). LDS padded to kill conflicts.
// Used for X ([C][S] per batch -> [S][C]) and all weight matrices.
// ---------------------------------------------------------------------------
__global__ __launch_bounds__(256) void k_trans(const float* __restrict__ in,
                                               _Float16* __restrict__ out,
                                               int K, int N) {
    __shared__ float T[32][33];
    const size_t bofs = (size_t)blockIdx.z * K * N;
    in  += bofs;
    out += bofs;
    const int n0 = blockIdx.x * 32, k0 = blockIdx.y * 32;
    const int tx = threadIdx.x, ty = threadIdx.y;
#pragma unroll
    for (int i = 0; i < 4; ++i)
        T[ty + 8 * i][tx] = in[(size_t)(k0 + ty + 8 * i) * N + n0 + tx];
    __syncthreads();
#pragma unroll
    for (int i = 0; i < 4; ++i)
        out[(size_t)(n0 + ty + 8 * i) * K + k0 + tx] =
            (_Float16)T[tx][ty + 8 * i];
}

// ---------------------------------------------------------------------------
// GEMM: A[M,K] f16 row-major  x  Bt[N,K] f16 (pre-transposed)  ->  C[M,N]
// MODE 0: f16 out row-major (scaled)
// MODE 1: f16 out with N = token dim m; out[(m/S)*VDIM + row][m%S] ([B][n][S])
//         (used to produce V^T with coalesced stores: A=WvT, Bt=xf)
// MODE 2: f32 out row-major (final output)
// block = 128 thr = 4 waves; wave tile 16x64; block tile 64x64
// A-frag per lane: k = 2i + 16*lh (16 contiguous)   -> uint4 at +0 and +8
// B-frag per lane: k = 2i + 16*lh (16 contiguous)   -> uint4 at +0 and +8
// ---------------------------------------------------------------------------
template <int MODE>
__global__ __launch_bounds__(128) void k_gemm(const _Float16* __restrict__ A,
                                              const _Float16* __restrict__ Bt,
                                              _Float16* __restrict__ outH,
                                              float* __restrict__ outF,
                                              int Mtot, int N, int Kdim,
                                              float scale) {
    const int lane = threadIdx.x & 31, wv = threadIdx.x >> 5;
    const int l16 = lane & 15, lh = lane >> 4;
    const int mtiles = Mtot >> 6;
    const int mb = blockIdx.x % mtiles, nb = blockIdx.x / mtiles;
    const int row0 = mb * 64 + wv * 16, col0 = nb * 64;

    v8f acc[4] = {};

    for (int kc = 0; kc < Kdim; kc += 32) {
        Frag16 a;
        const _Float16* ap = A + (size_t)(row0 + l16) * Kdim + kc + 16 * lh;
        a.q[0] = ldg4(ap);
        a.q[1] = ldg4(ap + 8);
#pragma unroll
        for (int t = 0; t < 4; ++t) {
            Frag16 bfr;
            const _Float16* bp =
                Bt + (size_t)(col0 + t * 16 + l16) * Kdim + kc + 16 * lh;
            bfr.q[0] = ldg4(bp);
            bfr.q[1] = ldg4(bp + 8);
            acc[t] = __builtin_amdgcn_wmma_f32_16x16x32_f16(
                false, a.h, false, bfr.h, (short)0, acc[t], false, false);
        }
    }

#pragma unroll
    for (int t = 0; t < 4; ++t)
#pragma unroll
        for (int v = 0; v < 8; ++v) {
            int row = row0 + v + 8 * lh;
            int col = col0 + t * 16 + l16;
            float val = acc[t][v] * scale;
            if (MODE == 0) {
                outH[(size_t)row * N + col] = (_Float16)val;
            } else if (MODE == 1) {   // V^T: col is token m; [B][row][s]
                int b = col >> 12, s = col & (SS - 1);
                outH[((size_t)b * VDIM + row) * SS + s] = (_Float16)val;
            } else {
                outF[(size_t)row * N + col] = val;
            }
        }
}

// ---------------------------------------------------------------------------
// One 64-key attention chunk, fully transposed formulation (no LDS).
//   L^T[key][q] = K_tile[16k x 32d] (A)  @  Q^T[32d x 16q] (B)
//   O^T[dv][q] += V^T[16dv x 32k] (A)    @  P^T[32k x 16q] (B)
// Each lane owns one query column q = qbase + l16; softmax state is scalar.
// Lane (l16, lh) holds keys kc + t*16 + v + 8*lh  (t tile, v vgpr).
// ---------------------------------------------------------------------------
template <bool MASKED>
DEV void attn_chunk(int kc, int qcol, int l16, int lh,
                    const _Float16* __restrict__ kBase,   // + row*KDIM
                    const _Float16* __restrict__ vBase,   // + col*SS
                    const Frag16& qb, v8f (&oaccT)[4], float& mrun,
                    float& srun) {
    // ---- transposed logits: rows = keys, cols = queries ----
    v8f clogT[4];
#pragma unroll
    for (int t = 0; t < 4; ++t) {
        Frag16 ka;   // A-frag: row = key kc+t*16+l16, contraction d
        const _Float16* kp =
            kBase + (size_t)(kc + t * 16 + l16) * KDIM + 16 * lh;
        ka.q[0] = ldg4(kp);
        ka.q[1] = ldg4(kp + 8);
        v8f z = {};
        clogT[t] = __builtin_amdgcn_wmma_f32_16x16x32_f16(
            false, ka.h, false, qb.h, (short)0, z, false, false);
    }
    if (MASKED) {
#pragma unroll
        for (int t = 0; t < 4; ++t)
#pragma unroll
            for (int v = 0; v < 8; ++v)
                clogT[t][v] =
                    (kc + t * 16 + v + 8 * lh > qcol) ? -1e9f : clogT[t][v];
    }

    // ---- chunk max: balanced tree + one half-swap ----
    float mc = fmaxf(fmaxf(max8(clogT[0]), max8(clogT[1])),
                     fmaxf(max8(clogT[2]), max8(clogT[3])));
    mc = fmaxf(mc, swap16(mc));
    const float mnew = fmaxf(mrun, mc);
    const float corr = __expf(mrun - mnew);
    mrun = mnew;

    // ---- exponentiate + rescale accumulators ----
    float e[4][8];
#pragma unroll
    for (int t = 0; t < 4; ++t)
#pragma unroll
        for (int v = 0; v < 8; ++v) {
            e[t][v] = __expf(clogT[t][v] - mnew);
            oaccT[t][v] *= corr;
        }

    // ---- build P^T B-frags in registers (keys 32s..32s+31 per frag) ----
    Frag16 pb[2];
    float psacc[2];
#pragma unroll
    for (int s = 0; s < 2; ++s) {
        float lo8[8], hi8[8];
#pragma unroll
        for (int v = 0; v < 8; ++v) {
            float send = lh ? e[2 * s][v] : e[2 * s + 1][v];
            float recv = swap16(send);
            lo8[v] = lh ? recv : e[2 * s][v];       // keys 32s + {0..7}+16lh..
            hi8[v] = lh ? e[2 * s + 1][v] : recv;   // keys 32s + {8..15}+16lh..
        }
#pragma unroll
        for (int i = 0; i < 4; ++i) {
            pb[s].u[i]     = pk2(lo8[2 * i], lo8[2 * i + 1]);
            pb[s].u[4 + i] = pk2(hi8[2 * i], hi8[2 * i + 1]);
        }
        psacc[s] = sum8(lo8) + sum8(hi8);
    }
    const float ps = psacc[0] + psacc[1];
    srun = srun * corr + (ps + swap16(ps));

    // ---- O^T += V^T @ P^T ----
#pragma unroll
    for (int t = 0; t < 4; ++t)
#pragma unroll
        for (int s = 0; s < 2; ++s) {
            Frag16 va;   // A-frag: row = dv t*16+l16, contraction = keys
            const _Float16* vp =
                vBase + (size_t)(t * 16 + l16) * SS + kc + 32 * s + 16 * lh;
            va.q[0] = ldg4(vp);
            va.q[1] = ldg4(vp + 8);
            oaccT[t] = __builtin_amdgcn_wmma_f32_16x16x32_f16(
                false, va.h, false, pb[s].h, (short)0, oaccT[t], false, false);
        }
}

// ---------------------------------------------------------------------------
// Flash attention. One wave per (b, head, 16-query tile). No LDS.
// Qp [M,KDIM] f16 (pre-scaled by dk^-0.5), Kp [M,KDIM] f16,
// Vt [B][VDIM][S] f16, Op [M,VDIM] f16 (merged heads).
// ---------------------------------------------------------------------------
__global__ __launch_bounds__(128) void k_attn(const _Float16* __restrict__ Qp,
                                              const _Float16* __restrict__ Kp,
                                              const _Float16* __restrict__ Vt,
                                              _Float16* __restrict__ Op) {
    const int lane = threadIdx.x & 31, wv = threadIdx.x >> 5;
    const int l16 = lane & 15, lh = lane >> 4;
    const int wid = blockIdx.x * 4 + wv;         // 0 .. B*NH*(S/16)-1
    const int qt = wid & 255;                    // S/16 = 256
    const int h  = (wid >> 8) & (NH - 1);
    const int b  = wid >> 11;
    const int qbase = qt * 16;
    const int qcol  = qbase + l16;               // this lane's query row

    const _Float16* kBase = Kp + (size_t)b * SS * KDIM + h * DK;
    const _Float16* vBase = Vt + ((size_t)b * VDIM + h * DV) * SS;

    // Q^T B-fragment: lane = query col, VGPR i holds d = 2i + 16*lh
    Frag16 qb;
    {
        const _Float16* qp =
            Qp + (size_t)(b * SS + qcol) * KDIM + h * DK + 16 * lh;
        qb.q[0] = ldg4(qp);
        qb.q[1] = ldg4(qp + 8);
    }

    float mrun = -1e30f, srun = 0.f;
    v8f oaccT[4] = {};

    // causal: keys 0..qbase+15.  qbase/64 mask-free chunks + 1 diagonal chunk.
    const int nfull = qbase >> 6;
    for (int ck = 0; ck < nfull; ++ck)
        attn_chunk<false>(ck * 64, qcol, l16, lh, kBase, vBase, qb, oaccT,
                          mrun, srun);
    attn_chunk<true>(nfull * 64, qcol, l16, lh, kBase, vBase, qb, oaccT, mrun,
                     srun);

    // ---- normalize + packed store: O^T cols are contiguous per lane ----
    const float rinv = 1.f / srun;
    _Float16* orow = Op + (size_t)(b * SS + qcol) * VDIM + h * DV + 8 * lh;
#pragma unroll
    for (int t = 0; t < 4; ++t) {
        uint4 w;
        w.x = pk2(oaccT[t][0] * rinv, oaccT[t][1] * rinv);
        w.y = pk2(oaccT[t][2] * rinv, oaccT[t][3] * rinv);
        w.z = pk2(oaccT[t][4] * rinv, oaccT[t][5] * rinv);
        w.w = pk2(oaccT[t][6] * rinv, oaccT[t][7] * rinv);
        *reinterpret_cast<uint4*>(orow + t * 16) = w;
    }
}

// ---------------------------------------------------------------------------
extern "C" void kernel_launch(void* const* d_in, const int* in_sizes, int n_in,
                              void* d_out, int out_size, void* d_ws,
                              size_t ws_size, hipStream_t stream) {
    const float* X  = (const float*)d_in[0];
    const float* Wq = (const float*)d_in[1];
    const float* Wk = (const float*)d_in[2];
    const float* Wv = (const float*)d_in[3];
    const float* Wo = (const float*)d_in[4];
    float* out = (float*)d_out;

    size_t off = 0;
    auto alloc = [&](size_t elems) {
        _Float16* p = (_Float16*)((char*)d_ws + off);
        off = (off + elems * sizeof(_Float16) + 255) & ~(size_t)255;
        return p;
    };
    _Float16* xf  = alloc((size_t)MM * CC);
    _Float16* WqT = alloc((size_t)KDIM * CC);
    _Float16* WkT = alloc((size_t)KDIM * CC);
    _Float16* WvT = alloc((size_t)VDIM * CC);
    _Float16* WoT = alloc((size_t)CC * VDIM);
    _Float16* Qb  = alloc((size_t)MM * KDIM);
    _Float16* Kb  = alloc((size_t)MM * KDIM);
    _Float16* Vt  = alloc((size_t)BB * VDIM * SS);
    _Float16* Ob  = alloc((size_t)MM * VDIM);

    // 1) coalesced tiled transposes (f32 -> f16)
    dim3 tb(32, 8);
    k_trans<<<dim3(SS / 32, CC / 32, BB), tb, 0, stream>>>(X, xf, CC, SS);
    k_trans<<<dim3(KDIM / 32, CC / 32, 1), tb, 0, stream>>>(Wq, WqT, CC, KDIM);
    k_trans<<<dim3(KDIM / 32, CC / 32, 1), tb, 0, stream>>>(Wk, WkT, CC, KDIM);
    k_trans<<<dim3(VDIM / 32, CC / 32, 1), tb, 0, stream>>>(Wv, WvT, CC, VDIM);
    k_trans<<<dim3(CC / 32, VDIM / 32, 1), tb, 0, stream>>>(Wo, WoT, VDIM, CC);

    // 2) projections (Q pre-scaled by dk^-0.5)
    const float qscale = 0.17677669529663687f;  // 1/sqrt(32)
    k_gemm<0><<<(MM / 64) * (KDIM / 64), 128, 0, stream>>>(
        xf, WqT, Qb, nullptr, MM, KDIM, CC, qscale);
    k_gemm<0><<<(MM / 64) * (KDIM / 64), 128, 0, stream>>>(
        xf, WkT, Kb, nullptr, MM, KDIM, CC, 1.f);
    // V^T computed as WvT[512,C] @ xf^T -> rows = dv, cols = tokens (coalesced)
    k_gemm<1><<<(VDIM / 64) * (MM / 64), 128, 0, stream>>>(
        WvT, xf, Vt, nullptr, VDIM, MM, CC, 1.f);

    // 3) causal flash attention: B*NH*(S/16) = 4096 waves
    k_attn<<<(BB * NH * (SS / 16)) / 4, 128, 0, stream>>>(Qb, Kb, Vt, Ob);

    // 4) output projection -> f32 d_out ([B,H,W,C] is contiguous [M,C])
    k_gemm<2><<<(MM / 64) * (CC / 64), 128, 0, stream>>>(
        Ob, WoT, nullptr, out, MM, CC, VDIM, 1.f);
}
